// MaskHead_50680614093435
// MI455X (gfx1250) — matrix operand order repaired and checked by reading, not compiled, not run on hardware
//
#include <hip/hip_runtime.h>
#include <math.h>

typedef __attribute__((ext_vector_type(16))) _Float16 v16h;
typedef __attribute__((ext_vector_type(8)))  _Float16 v8h;
typedef __attribute__((ext_vector_type(8)))  float    v8f;

#define NBOX 2048
#define RMAX 300
#define SS   14
#define PS   16    // padded side (1-px halo)
#define PAREA 256  // 16*16
#define OHW  26
#define C1   1792
#define NPIX 196   // 14*14
#define NPIX2 676  // 26*26

// ---------------------------------------------------------------------------
// 1) NMS: single 1024-thread block reproduces _nms_select for B=1.
// ---------------------------------------------------------------------------
__global__ __launch_bounds__(1024) void nms_kernel(const float* __restrict__ yolo,
                                                   float* __restrict__ rboxes)
{
    __shared__ float sob[NBOX][4];
    __shared__ float ssc[NBOX];
    __shared__ int   sidx[NBOX];
    __shared__ float soff[NBOX];
    __shared__ unsigned char skeep[NBOX];
    __shared__ float smax;
    const int tid = threadIdx.x;

    float mx = -1e30f;
    for (int i = tid; i < NBOX; i += 1024) {
        const float* row = yolo + i * 85;
        mx = fmaxf(mx, fmaxf(fmaxf(row[0], row[1]), fmaxf(row[2], row[3])));
    }
    ssc[tid] = mx;
    __syncthreads();
    for (int s = 512; s > 0; s >>= 1) {
        if (tid < s) ssc[tid] = fmaxf(ssc[tid], ssc[tid + s]);
        __syncthreads();
    }
    if (tid == 0) smax = ssc[0];
    __syncthreads();
    const float offscale = smax + 1.0f;

    for (int i = tid; i < NBOX; i += 1024) {
        const float* row = yolo + i * 85;
        float best = row[5]; int bc = 0;
        for (int c = 1; c < 80; ++c) { float v = row[5 + c]; if (v > best) { best = v; bc = c; } }
        ssc[i]  = row[4];
        sidx[i] = i;
        soff[i] = (float)bc * offscale;
    }
    __syncthreads();

    // bitonic sort: descending by score
    for (int k = 2; k <= NBOX; k <<= 1) {
        for (int j = k >> 1; j > 0; j >>= 1) {
            __syncthreads();
            for (int t = tid; t < NBOX; t += 1024) {
                int l = t ^ j;
                if (l > t) {
                    bool descB = ((t & k) == 0);
                    float a = ssc[t], b = ssc[l];
                    bool sw = descB ? (a < b) : (a > b);
                    if (sw) {
                        ssc[t] = b; ssc[l] = a;
                        int ti = sidx[t]; sidx[t] = sidx[l]; sidx[l] = ti;
                    }
                }
            }
        }
    }
    __syncthreads();

    for (int i = tid; i < NBOX; i += 1024) {
        int o = sidx[i];
        const float* row = yolo + o * 85;
        float off = soff[o];
        sob[i][0] = row[0] + off; sob[i][1] = row[1] + off;
        sob[i][2] = row[2] + off; sob[i][3] = row[3] + off;
        skeep[i] = 1;
    }
    __syncthreads();

    for (int i = 0; i < NBOX - 1; ++i) {
        __syncthreads();
        if (!skeep[i]) continue;
        float bx0 = sob[i][0], by0 = sob[i][1], bx1 = sob[i][2], by1 = sob[i][3];
        float ai = (bx1 - bx0) * (by1 - by0);
        for (int jx = i + 1 + tid; jx < NBOX; jx += 1024) {
            float cx0 = sob[jx][0], cy0 = sob[jx][1], cx1 = sob[jx][2], cy1 = sob[jx][3];
            float aj = (cx1 - cx0) * (cy1 - cy0);
            float ix0 = fmaxf(bx0, cx0), iy0 = fmaxf(by0, cy0);
            float ix1 = fminf(bx1, cx1), iy1 = fminf(by1, cy1);
            float iw = fmaxf(ix1 - ix0, 0.f), ih = fmaxf(iy1 - iy0, 0.f);
            float inter = iw * ih;
            float iou = inter / (ai + aj - inter + 1e-9f);
            if (iou > 0.5f) skeep[jx] = 0;
        }
    }
    __syncthreads();

    if (tid == 0) {
        int cnt = 0;
        for (int pass = 0; pass < 2 && cnt < RMAX; ++pass) {
            for (int j = 0; j < NBOX && cnt < RMAX; ++j) {
                bool take = pass == 0 ? (skeep[j] != 0) : (skeep[j] == 0);
                if (take) {
                    const float* row = yolo + sidx[j] * 85;
                    rboxes[cnt * 4 + 0] = row[0]; rboxes[cnt * 4 + 1] = row[1];
                    rboxes[cnt * 4 + 2] = row[2]; rboxes[cnt * 4 + 3] = row[3];
                    ++cnt;
                }
            }
        }
    }
}

// ---------------------------------------------------------------------------
// 2) ROI align into PIXEL-MAJOR padded f16 layout ft[r][16*16][C1]; halo = 0.
//    Thread order has channel fastest -> coalesced writes.
// ---------------------------------------------------------------------------
__global__ void roi_align_kernel(const float* __restrict__ f0, const float* __restrict__ f1,
                                 const float* __restrict__ f2, const float* __restrict__ rboxes,
                                 _Float16* __restrict__ ft)
{
    long long t = (long long)blockIdx.x * blockDim.x + threadIdx.x;
    const long long total = (long long)RMAX * C1 * PAREA;
    if (t >= total) return;
    int c   = (int)(t % C1);
    int pix = (int)((t / C1) & 255);
    int r   = (int)(t / ((long long)C1 * PAREA));
    int pj = pix & 15, pi = pix >> 4;

    if (pi == 0 || pi == 15 || pj == 0 || pj == 15) { ft[t] = (_Float16)0.f; return; }
    int i = pi - 1, j = pj - 1;

    const float* f; int H, W, ch; float scale;
    if (c < 256)      { f = f0; H = 64; W = 64; ch = c;       scale = 0.125f;   }
    else if (c < 768) { f = f1; H = 32; W = 32; ch = c - 256; scale = 0.0625f;  }
    else              { f = f2; H = 16; W = 16; ch = c - 768; scale = 0.03125f; }

    float bx0 = rboxes[r * 4 + 0] * scale - 0.5f;
    float by0 = rboxes[r * 4 + 1] * scale - 0.5f;
    float bx1 = rboxes[r * 4 + 2] * scale - 0.5f;
    float by1 = rboxes[r * 4 + 3] * scale - 0.5f;
    float bw = (bx1 - bx0) * (1.0f / SS), bh = (by1 - by0) * (1.0f / SS);
    float xs = bx0 + ((float)j + 0.5f) * bw;
    float ys = by0 + ((float)i + 0.5f) * bh;
    bool valid = (ys >= -1.f) && (ys <= (float)H) && (xs >= -1.f) && (xs <= (float)W);
    float y = fminf(fmaxf(ys, 0.f), (float)(H - 1));
    float x = fminf(fmaxf(xs, 0.f), (float)(W - 1));
    int y0 = min(max((int)floorf(y), 0), H - 1);
    int x0 = min(max((int)floorf(x), 0), W - 1);
    int y1 = min(y0 + 1, H - 1), x1 = min(x0 + 1, W - 1);
    float ly = y - (float)y0, lx = x - (float)x0;
    float hy = 1.f - ly, hx = 1.f - lx;
    const float* fc = f + (long long)ch * H * W;
    float v = hy * hx * fc[y0 * W + x0] + hy * lx * fc[y0 * W + x1]
            + ly * hx * fc[y1 * W + x0] + ly * lx * fc[y1 * W + x1];
    ft[t] = (_Float16)(valid ? v : 0.f);
}

// ---------------------------------------------------------------------------
// 3) Weight packing into WMMA A-fragment layout (f16 16x32):
//    lane<16: halves 0-15 -> K 0-15 (row M = lane)
//    lane>=16: halves 0-15 -> K 16-31
// K ordering is POSITION-MAJOR (ic innermost):
//   mode 0 conv3x3: k = (ky*3+kx)*Cin + ic, src [Cout][Cin][3][3]
//   mode 1 deconv : k = (ky*2+kx)*256 + ic, src wt[ic][oc][1-ky][1-kx]
//   mode 2 conv1x1: k = ic, src [Cout][Cin]
// ---------------------------------------------------------------------------
__global__ void pack_kernel(const float* __restrict__ src, _Float16* __restrict__ dst,
                            int K, int Cin, int numOcT, int mode)
{
    int t = blockIdx.x * blockDim.x + threadIdx.x;
    int total = (K >> 5) * numOcT * 512;
    if (t >= total) return;
    int h    = t & 15;
    int lane = (t >> 4) & 31;
    int ot   = (t >> 9) % numOcT;
    int ks   = t / (512 * numOcT);
    int M    = ot * 16 + (lane & 15);
    int kl   = h + ((lane < 16) ? 0 : 16);
    int k    = ks * 32 + kl;
    float v;
    if (mode == 0) {
        int kk = k / Cin, ic = k - kk * Cin;
        v = src[(M * Cin + ic) * 9 + kk];
    } else if (mode == 1) {
        int kp = k >> 8, ic = k & 255;
        v = src[((ic * 256 + M) * 2 + (1 - (kp >> 1))) * 2 + (1 - (kp & 1))];
    } else {
        v = src[M * Cin + k];
    }
    dst[t] = (_Float16)v;
}

__global__ void bnprep_kernel(const float* __restrict__ g, const float* __restrict__ b,
                              const float* __restrict__ m, const float* __restrict__ v,
                              int C, float* __restrict__ scale, float* __restrict__ bias)
{
    int c = blockIdx.x * blockDim.x + threadIdx.x;
    if (c >= C) return;
    float inv = g[c] * rsqrtf(v[c] + 1e-5f);
    scale[c] = inv;
    bias[c]  = b[c] - m[c] * inv;
}

__global__ void zero_kernel(uint4* __restrict__ p, int n16)
{
    int i = blockIdx.x * blockDim.x + threadIdx.x;
    if (i < n16) { uint4 z = {0u, 0u, 0u, 0u}; p[i] = z; }
}

// ---------------------------------------------------------------------------
// 4) 3x3 conv implicit GEMM, PIXEL-MAJOR activations in[pix][Cin] (padded
//    16x16 pixel grid), out[pix][256]. One wave: 64 outch x 16 pixels, 4 A
//    fragments share one contiguously-loaded v16h B fragment.
//    grid = (RMAX, 4, 13), block = 32.
// ---------------------------------------------------------------------------
__global__ __launch_bounds__(32) void conv3x3_wmma(const _Float16* __restrict__ in,
                                                   _Float16* __restrict__ out,
                                                   const _Float16* __restrict__ pw,
                                                   const float* __restrict__ scale,
                                                   const float* __restrict__ bias,
                                                   int Cin)
{
    const int roi = blockIdx.x, otb = blockIdx.y, pt = blockIdx.z;
    const int lane = threadIdx.x;
    const int n = lane & 15;
    const int p = pt * 16 + n;
    const bool pvalid = p < NPIX;
    const int pc = pvalid ? p : 0;
    const int oy = pc / SS, ox = pc - oy * SS;
    const int klaneoff = (lane < 16) ? 0 : 16;
    const int icn = Cin >> 5;                       // 32-chunks per tap position
    const _Float16* inR = in + (long long)roi * PAREA * Cin;
    const v16h* pA = (const v16h*)pw;

    v8f acc0 = {}, acc1 = {}, acc2 = {}, acc3 = {};
#pragma unroll
    for (int kk = 0; kk < 9; ++kk) {
        const int ky = kk / 3, kx = kk - ky * 3;
        // tap pixel (padded): (oy+ky, ox+kx); channel run contiguous per lane
        const _Float16* bbase = inR + (size_t)((oy + ky) * PS + ox + kx) * Cin + klaneoff;
        for (int icb = 0; icb < icn; ++icb) {
            const int ks = kk * icn + icb;
            const v16h* ap = pA + (size_t)(ks * 16 + otb * 4) * 32 + lane;
            v16h a0 = ap[0], a1 = ap[32], a2 = ap[64], a3 = ap[96];
            v16h b = *(const v16h*)(bbase + icb * 32);
            acc0 = __builtin_amdgcn_wmma_f32_16x16x32_f16(false, a0, false, b, (short)0, acc0, false, false);
            acc1 = __builtin_amdgcn_wmma_f32_16x16x32_f16(false, a1, false, b, (short)0, acc1, false, false);
            acc2 = __builtin_amdgcn_wmma_f32_16x16x32_f16(false, a2, false, b, (short)0, acc2, false, false);
            acc3 = __builtin_amdgcn_wmma_f32_16x16x32_f16(false, a3, false, b, (short)0, acc3, false, false);
        }
    }
    if (!pvalid) return;
    const int ppad = (oy + 1) * PS + (ox + 1);
    _Float16* outP = out + ((long long)roi * PAREA + ppad) * 256;
    const int mhalf = (lane < 16) ? 0 : 8;
#pragma unroll
    for (int mi = 0; mi < 4; ++mi) {
        const v8f acc = mi == 0 ? acc0 : mi == 1 ? acc1 : mi == 2 ? acc2 : acc3;
        const int mbase = (otb * 4 + mi) * 16 + mhalf;
        v8h h8;
#pragma unroll
        for (int r8 = 0; r8 < 8; ++r8) {
            float v = acc[r8] * scale[mbase + r8] + bias[mbase + r8];
            h8[r8] = (_Float16)fmaxf(v, 0.f);
        }
        *(v8h*)(outP + mbase) = h8;   // 16B packed store of 8 consecutive oc
    }
}

// ---------------------------------------------------------------------------
// 5) transposed 2x2 stride-2 conv (pad 1 -> 26x26), parity-tiled output so
//    only the matching kp block of K contributes (effective K = 256).
//    in[pix(16x16 padded)][256], out x4[pix(26x26)][256].
//    grid = (RMAX, 4, 4*11), block = 32.
// ---------------------------------------------------------------------------
__global__ __launch_bounds__(32) void deconv_wmma(const _Float16* __restrict__ in,
                                                  _Float16* __restrict__ out,
                                                  const _Float16* __restrict__ pw,
                                                  const float* __restrict__ scale,
                                                  const float* __restrict__ bias)
{
    const int roi = blockIdx.x, otb = blockIdx.y;
    const int parity = blockIdx.z / 11, tile = blockIdx.z - parity * 11;
    const int py = parity >> 1, px = parity & 1;
    const int lane = threadIdx.x;
    const int n = lane & 15;
    const int q = tile * 16 + n;          // index in 13x13 parity subgrid
    const bool qvalid = q < 169;
    const int qc = qvalid ? q : 0;
    const int sy = qc / 13, sx = qc - sy * 13;
    const int iy = sy + py, ix = sx + px;             // input tap (0..13)
    const int klaneoff = (lane < 16) ? 0 : 16;
    const _Float16* bbase = in + ((long long)roi * PAREA + (iy + 1) * PS + ix + 1) * 256 + klaneoff;
    const v16h* pA = (const v16h*)pw;

    v8f acc0 = {}, acc1 = {}, acc2 = {}, acc3 = {};
#pragma unroll
    for (int icb = 0; icb < 8; ++icb) {
        const int ks = parity * 8 + icb;              // K = kp*256 + ic ordering
        const v16h* ap = pA + (size_t)(ks * 16 + otb * 4) * 32 + lane;
        v16h a0 = ap[0], a1 = ap[32], a2 = ap[64], a3 = ap[96];
        v16h b = *(const v16h*)(bbase + icb * 32);
        acc0 = __builtin_amdgcn_wmma_f32_16x16x32_f16(false, a0, false, b, (short)0, acc0, false, false);
        acc1 = __builtin_amdgcn_wmma_f32_16x16x32_f16(false, a1, false, b, (short)0, acc1, false, false);
        acc2 = __builtin_amdgcn_wmma_f32_16x16x32_f16(false, a2, false, b, (short)0, acc2, false, false);
        acc3 = __builtin_amdgcn_wmma_f32_16x16x32_f16(false, a3, false, b, (short)0, acc3, false, false);
    }
    if (!qvalid) return;
    const int oy = 2 * sy + py, ox = 2 * sx + px;
    _Float16* outP = out + ((long long)roi * NPIX2 + oy * OHW + ox) * 256;
    const int mhalf = (lane < 16) ? 0 : 8;
#pragma unroll
    for (int mi = 0; mi < 4; ++mi) {
        const v8f acc = mi == 0 ? acc0 : mi == 1 ? acc1 : mi == 2 ? acc2 : acc3;
        const int mbase = (otb * 4 + mi) * 16 + mhalf;
        v8h h8;
#pragma unroll
        for (int r8 = 0; r8 < 8; ++r8) {
            float v = acc[r8] * scale[mbase + r8] + bias[mbase + r8];
            h8[r8] = (_Float16)fmaxf(v, 0.f);
        }
        *(v8h*)(outP + mbase) = h8;
    }
}

// ---------------------------------------------------------------------------
// 6) 1x1 conv (256->80) + BN + sigmoid -> float masks (NCHW, reference
//    layout). in x4[pix][256]. One wave: all 80 outch x 16 pixels.
//    grid = (RMAX, 1, 43), block = 32.
// ---------------------------------------------------------------------------
__global__ __launch_bounds__(32) void conv1x1_sigmoid_wmma(const _Float16* __restrict__ in,
                                                           float* __restrict__ outMasks,
                                                           const _Float16* __restrict__ pw,
                                                           const float* __restrict__ scale,
                                                           const float* __restrict__ bias)
{
    const int roi = blockIdx.x, pt = blockIdx.z;
    const int lane = threadIdx.x;
    const int n = lane & 15;
    const int p = pt * 16 + n;
    const bool pvalid = p < NPIX2;
    const int pc = pvalid ? p : 0;
    const int klaneoff = (lane < 16) ? 0 : 16;
    const _Float16* bbase = in + ((long long)roi * NPIX2 + pc) * 256 + klaneoff;
    const v16h* pA = (const v16h*)pw;

    v8f acc0 = {}, acc1 = {}, acc2 = {}, acc3 = {}, acc4 = {};
#pragma unroll
    for (int ks = 0; ks < 8; ++ks) {
        const v16h* ap = pA + (size_t)(ks * 5) * 32 + lane;
        v16h a0 = ap[0], a1 = ap[32], a2 = ap[64], a3 = ap[96], a4 = ap[128];
        v16h b = *(const v16h*)(bbase + ks * 32);
        acc0 = __builtin_amdgcn_wmma_f32_16x16x32_f16(false, a0, false, b, (short)0, acc0, false, false);
        acc1 = __builtin_amdgcn_wmma_f32_16x16x32_f16(false, a1, false, b, (short)0, acc1, false, false);
        acc2 = __builtin_amdgcn_wmma_f32_16x16x32_f16(false, a2, false, b, (short)0, acc2, false, false);
        acc3 = __builtin_amdgcn_wmma_f32_16x16x32_f16(false, a3, false, b, (short)0, acc3, false, false);
        acc4 = __builtin_amdgcn_wmma_f32_16x16x32_f16(false, a4, false, b, (short)0, acc4, false, false);
    }
    if (!pvalid) return;
    float* outR = outMasks + (long long)roi * 80 * NPIX2;
    const int mhalf = (lane < 16) ? 0 : 8;
#pragma unroll
    for (int mi = 0; mi < 5; ++mi) {
        const v8f acc = mi == 0 ? acc0 : mi == 1 ? acc1 : mi == 2 ? acc2 : mi == 3 ? acc3 : acc4;
        const int mbase = mi * 16 + mhalf;
#pragma unroll
        for (int r8 = 0; r8 < 8; ++r8) {
            int oc = mbase + r8;
            float v = acc[r8] * scale[oc] + bias[oc];
            outR[oc * NPIX2 + p] = 1.f / (1.f + __expf(-v));
        }
    }
}

__global__ void copy_kernel(const float* __restrict__ src, float* __restrict__ dst, int n)
{
    int i = blockIdx.x * blockDim.x + threadIdx.x;
    if (i < n) dst[i] = src[i];
}

// ---------------------------------------------------------------------------
// Host. d_in order (setup_inputs dict insertion order):
// 0 yolo, 1 feat0, 2 feat1, 3 feat2, 4 w1, 5 w2, 6 w3, 7 wt, 8 w5,
// then g,b,m,v for layers 1..5 (indices 9..28).
// ---------------------------------------------------------------------------
extern "C" void kernel_launch(void* const* d_in, const int* in_sizes, int n_in,
                              void* d_out, int out_size, void* d_ws, size_t ws_size,
                              hipStream_t stream)
{
    (void)in_sizes; (void)n_in; (void)out_size; (void)ws_size;
    const float* yolo = (const float*)d_in[0];
    const float* f0   = (const float*)d_in[1];
    const float* f1   = (const float*)d_in[2];
    const float* f2   = (const float*)d_in[3];
    const float* w1   = (const float*)d_in[4];
    const float* w2   = (const float*)d_in[5];
    const float* w3   = (const float*)d_in[6];
    const float* wt   = (const float*)d_in[7];
    const float* w5   = (const float*)d_in[8];
    const float* bn[5][4];
    for (int l = 0; l < 5; ++l)
        for (int q = 0; q < 4; ++q)
            bn[l][q] = (const float*)d_in[9 + l * 4 + q]; // g,b,m,v

    char* ws = (char*)d_ws;
    size_t off = 0;
    auto alloc = [&](size_t bytes) -> void* {
        void* p = ws + off;
        off += (bytes + 255) & ~(size_t)255;
        return p;
    };
    const size_t ftBytes = (size_t)RMAX * C1 * PAREA * 2;   // padded, pixel-major
    const size_t xBytes  = (size_t)RMAX * 256 * PAREA * 2;  // padded, pixel-major
    _Float16* ft  = (_Float16*)alloc(ftBytes);
    _Float16* xA  = (_Float16*)alloc(xBytes);
    _Float16* xB  = (_Float16*)alloc(xBytes);
    _Float16* x4  = (_Float16*)alloc((size_t)RMAX * 256 * NPIX2 * 2); // pixel-major
    _Float16* pw1 = (_Float16*)alloc((size_t)256 * 16128 * 2);
    _Float16* pw2 = (_Float16*)alloc((size_t)256 * 2304 * 2);
    _Float16* pw3 = (_Float16*)alloc((size_t)256 * 2304 * 2);
    _Float16* pwt = (_Float16*)alloc((size_t)256 * 1024 * 2);
    _Float16* pw5 = (_Float16*)alloc((size_t)80 * 256 * 2);
    float* sc[5]; float* bi[5];
    const int bnC[5] = {256, 256, 256, 256, 80};
    for (int l = 0; l < 5; ++l) { sc[l] = (float*)alloc(bnC[l] * 4); bi[l] = (float*)alloc(bnC[l] * 4); }
    float* rboxes = (float*)alloc((size_t)RMAX * 4 * 4);

    // NMS -> 300 boxes
    nms_kernel<<<1, 1024, 0, stream>>>(yolo, rboxes);

    // BN folding
    for (int l = 0; l < 5; ++l)
        bnprep_kernel<<<(bnC[l] + 255) / 256, 256, 0, stream>>>(bn[l][0], bn[l][1], bn[l][2], bn[l][3],
                                                                bnC[l], sc[l], bi[l]);
    // Weight packing (position-major K, ic innermost)
    pack_kernel<<<(504 * 16 * 512) / 256, 256, 0, stream>>>(w1, pw1, 16128, 1792, 16, 0);
    pack_kernel<<<(72  * 16 * 512) / 256, 256, 0, stream>>>(w2, pw2, 2304, 256, 16, 0);
    pack_kernel<<<(72  * 16 * 512) / 256, 256, 0, stream>>>(w3, pw3, 2304, 256, 16, 0);
    pack_kernel<<<(32  * 16 * 512) / 256, 256, 0, stream>>>(wt, pwt, 1024, 256, 16, 1);
    pack_kernel<<<(8   *  5 * 512) / 256, 256, 0, stream>>>(w5, pw5, 256, 256, 5, 2);

    // Zero padded activation buffers (halo rows must be 0 every call)
    {
        int n16 = (int)(xBytes / 16);
        zero_kernel<<<(n16 + 255) / 256, 256, 0, stream>>>((uint4*)xA, n16);
        zero_kernel<<<(n16 + 255) / 256, 256, 0, stream>>>((uint4*)xB, n16);
    }

    // ROI align -> padded pixel-major ft (f16)
    {
        long long total = (long long)RMAX * C1 * PAREA;
        roi_align_kernel<<<(unsigned)((total + 255) / 256), 256, 0, stream>>>(f0, f1, f2, rboxes, ft);
    }

    // mask head
    conv3x3_wmma<<<dim3(RMAX, 4, 13), 32, 0, stream>>>(ft, xA, pw1, sc[0], bi[0], 1792);
    conv3x3_wmma<<<dim3(RMAX, 4, 13), 32, 0, stream>>>(xA, xB, pw2, sc[1], bi[1], 256);
    conv3x3_wmma<<<dim3(RMAX, 4, 13), 32, 0, stream>>>(xB, xA, pw3, sc[2], bi[2], 256);
    deconv_wmma<<<dim3(RMAX, 4, 44), 32, 0, stream>>>(xA, x4, pwt, sc[3], bi[3]);
    conv1x1_sigmoid_wmma<<<dim3(RMAX, 1, 43), 32, 0, stream>>>(x4, (float*)d_out + 2048 * 85,
                                                               pw5, sc[4], bi[4]);
    // yolo passthrough (first tuple element)
    copy_kernel<<<(2048 * 85 + 255) / 256, 256, 0, stream>>>(yolo, (float*)d_out, 2048 * 85);
}
